// SDCN_22204980920355
// MI455X (gfx1250) — matrix operand order repaired and checked by reading, not compiled, not run on hardware
//
// SDCN forward for MI455X (gfx1250, wave32, WMMA).
//
// Roofline: ~1 TFLOP of GEMM work vs ~1.5-3 GB of adjacency traffic at
// 23.3 TB/s -> memory bound on the 21 "adj @ (feat @ W)" propagations
// (each streams a 144 MB f32 adjacency; the 192 MB L2 holds one adj whole).
// Strategy: stream f32 adjacency, convert to f16 (scaled by 2^13 to stay out
// of the f16 subnormal range; 2^-13 folded into the epilogue alpha) while
// staging into LDS, and run the MACs through v_wmma_f32_16x16x32_f16 with
// f32 accumulation. Every WMMA GEMM can emit both an f32 result (gates /
// outputs) and an f16 result (operand of the next GEMM) in one pass.
// All f16 operands use 8-element-padded, zero-filled leading dimensions, so
// the K-loop body is pure vector staging (b128/b64 loads, b32 LDS stores)
// with row indices clamped instead of guarded; the single ragged-K tail
// lives outside the loop.

#include <hip/hip_runtime.h>

typedef __attribute__((ext_vector_type(16))) _Float16 v16h;
typedef __attribute__((ext_vector_type(8)))  _Float16 v8h;
typedef __attribute__((ext_vector_type(4)))  _Float16 v4h;
typedef __attribute__((ext_vector_type(2)))  _Float16 v2h;
typedef __attribute__((ext_vector_type(8)))  float    v8f;
typedef __attribute__((ext_vector_type(4)))  float    v4f;

#define BM 128
#define BN 64
#define BK 32
#define LDK (BK + 8)   // LDS row stride in halves: 80B, keeps 16B alignment

static_assert(sizeof(_Float16) == 2, "f16 size");

__device__ __forceinline__ int imin(int a, int b) { return a < b ? a : b; }

__device__ __forceinline__ v16h cat16(v8h lo, v8h hi) {
  v16h r;
#pragma unroll
  for (int i = 0; i < 8; ++i) { r[i] = lo[i]; r[i + 8] = hi[i]; }
  return r;
}

// ---------------------------------------------------------------------------
// Tiled WMMA GEMM:  C[M,N] = act(alpha * (A[M,K] @ B[K,N]) + bias)
//   A16: A is _Float16 row-major, REQUIRES lda % 8 == 0; else A is float,
//        REQUIRES lda % 4 == 0, converted to f16 (* ascale) while staging.
//   B: _Float16 row-major K x N, REQUIRES ldb % 4 == 0 with cols [N, ldb)
//      zero-filled (all producers in this file guarantee this).
//   Cf (f32, stride N) and/or Ch (f16, stride ldch, pad zero-filled) written
//   if non-null. relu applied if nonzero.
// Block: 256 threads = 8 waves (4 along M x 2 along N); each wave owns a
// 32x32 macro-tile = 2x2 v_wmma_f32_16x16x32_f16 tiles.
// Main K loop is vector-only (row index clamped to M-1; out-of-range rows
// compute garbage that the epilogue never stores); the ragged K tail is a
// single guarded block outside the loop.
// ---------------------------------------------------------------------------
template <bool A16>
__global__ __launch_bounds__(256) void gemm_wmma(
    const void* __restrict__ Av, int lda, const _Float16* __restrict__ B,
    int ldb, const float* __restrict__ bias, float* __restrict__ Cf,
    _Float16* __restrict__ Ch, int ldch, int M, int N, int K, float alpha,
    float ascale, int relu)
{
  __shared__ _Float16 As[BM][LDK];
  __shared__ _Float16 Bs[BN][LDK];   // transposed: Bs[n][k]

  const int tid  = threadIdx.x;
  const int m0   = blockIdx.x * BM;
  const int n0   = blockIdx.y * BN;
  const int wave = tid >> 5;
  const int lane = tid & 31;
  const int wm   = wave & 3;    // wave row 0..3  (32 rows each)
  const int wn   = wave >> 2;   // wave col 0..1  (32 cols each)
  const int lrow = lane & 15;
  const int hi16 = lane >> 4;   // 0: lanes 0-15, 1: lanes 16-31

  // A staging: 256 threads, each stages 16 halves of one row.
  const int arow = tid >> 1;
  const int acol = (tid & 1) << 4;
  // B staging: thread stages a 2(K) x 4(N) sub-tile: b64 loads from two
  // consecutive K rows, packed into transposed b32 LDS stores.
  const int brp  = tid >> 4;          // K row-pair 0..15
  const int bcg  = (tid & 15) << 2;   // col group 0,4,...,60

  const int  gm   = m0 + arow;
  const int  gmc  = imin(gm, M - 1);      // clamped row (branchless staging)
  const bool bok  = (n0 + bcg) < ldb;     // B pad cols are zero-filled
  const v4h  vz4  = {};

  v8f acc[2][2] = {};

  // fragment construction + 2x2 WMMA from the staged LDS tiles
  auto mma_step = [&]() {
    // A (16-bit 16x32): lane row = lane%16; holds K=ka..ka+7 and
    // K=ka+16..ka+23, ka = (lane/16)*8.
    const int ka = hi16 * 8;
    // B: lane col = lane%16; holds K=kb..kb+7 and K=kb+16..kb+23, kb = (lane/16)*8.
    const int kb = hi16 * 8;
    v16h af[2], bf[2];
#pragma unroll
    for (int i = 0; i < 2; ++i) {
      const _Float16* p = &As[wm * 32 + i * 16 + lrow][ka];
      af[i] = cat16(*(const v8h*)p, *(const v8h*)(p + 16));
    }
#pragma unroll
    for (int j = 0; j < 2; ++j) {
      const _Float16* p = &Bs[wn * 32 + j * 16 + lrow][kb];
      bf[j] = cat16(*(const v8h*)p, *(const v8h*)(p + 16));
    }
#pragma unroll
    for (int i = 0; i < 2; ++i)
#pragma unroll
      for (int j = 0; j < 2; ++j)
        acc[i][j] = __builtin_amdgcn_wmma_f32_16x16x32_f16(
            false, af[i], false, bf[j], (short)0, acc[i][j], false, false);
  };

  const int kFull = K - (K % BK);

  for (int k0 = 0; k0 < kFull; k0 += BK) {
    // ---- stage A tile: pure vector, row-clamped ----
    if (A16) {
      const _Float16* A = (const _Float16*)Av;
      const v8h* p = (const v8h*)(A + (size_t)gmc * lda + k0 + acol);
      *(v8h*)&As[arow][acol]     = p[0];
      *(v8h*)&As[arow][acol + 8] = p[1];
      if (k0 + BK < K)
        __builtin_prefetch(A + (size_t)gmc * lda + k0 + BK + acol, 0, 1);
    } else {
      const float* A = (const float*)Av;
      const v4f* p = (const v4f*)(A + (size_t)gmc * lda + k0 + acol);
      const v4f f0 = p[0], f1 = p[1], f2 = p[2], f3 = p[3];
      v8h h0, h1;
#pragma unroll
      for (int e = 0; e < 4; ++e) {
        h0[e]     = (_Float16)(f0[e] * ascale);
        h0[e + 4] = (_Float16)(f1[e] * ascale);
        h1[e]     = (_Float16)(f2[e] * ascale);
        h1[e + 4] = (_Float16)(f3[e] * ascale);
      }
      *(v8h*)&As[arow][acol]     = h0;
      *(v8h*)&As[arow][acol + 8] = h1;
      if (k0 + BK < K)
        __builtin_prefetch(A + (size_t)gmc * lda + k0 + BK + acol, 0, 1);
    }
    // ---- stage B tile: pure vector (rows k0..k0+31 all < K here) ----
    {
      v4h b0 = vz4, b1 = vz4;
      if (bok) {
        const _Float16* Bp = B + (size_t)(k0 + brp * 2) * ldb + n0 + bcg;
        b0 = *(const v4h*)Bp;
        b1 = *(const v4h*)(Bp + ldb);
      }
#pragma unroll
      for (int i = 0; i < 4; ++i) {
        v2h pr; pr[0] = b0[i]; pr[1] = b1[i];
        *(v2h*)&Bs[bcg + i][brp * 2] = pr;
      }
    }
    __syncthreads();
    mma_step();
    __syncthreads();
  }

  // ---- ragged K tail (at most one partial step) ----
  if (kFull < K) {
    if (A16) {
      const _Float16* A = (const _Float16*)Av;
#pragma unroll
      for (int j = 0; j < 16; ++j) {
        const int gk = kFull + acol + j;
        const _Float16 v = A[(size_t)gmc * lda + imin(gk, K - 1)];
        As[arow][acol + j] = (gk < K) ? v : (_Float16)0.0f;
      }
    } else {
      const float* A = (const float*)Av;
#pragma unroll
      for (int j = 0; j < 16; ++j) {
        const int gk = kFull + acol + j;
        const float v = A[(size_t)gmc * lda + imin(gk, K - 1)];
        As[arow][acol + j] =
            (gk < K) ? (_Float16)(v * ascale) : (_Float16)0.0f;
      }
    }
    {
      const int gk0 = kFull + brp * 2;
      const int gk1 = gk0 + 1;
      v4h b0 = vz4, b1 = vz4;
      if (bok) {
        b0 = *(const v4h*)(B + (size_t)imin(gk0, K - 1) * ldb + n0 + bcg);
        b1 = *(const v4h*)(B + (size_t)imin(gk1, K - 1) * ldb + n0 + bcg);
        if (gk0 >= K) b0 = vz4;
        if (gk1 >= K) b1 = vz4;
      }
#pragma unroll
      for (int i = 0; i < 4; ++i) {
        v2h pr; pr[0] = b0[i]; pr[1] = b1[i];
        *(v2h*)&Bs[bcg + i][brp * 2] = pr;
      }
    }
    __syncthreads();
    mma_step();
    __syncthreads();
  }

  // ---- epilogue: C layout = VGPR r, lanes 0-15 M=r, lanes 16-31 M=8+r ----
#pragma unroll
  for (int i = 0; i < 2; ++i) {
#pragma unroll
    for (int j = 0; j < 2; ++j) {
      const int col   = n0 + wn * 32 + j * 16 + lrow;
      const int rbase = m0 + wm * 32 + i * 16 + hi16 * 8;
      const float bv  = (bias != nullptr && col < N) ? bias[col] : 0.0f;
#pragma unroll
      for (int r = 0; r < 8; ++r) {
        const int row = rbase + r;
        if (row < M) {
          if (col < N) {
            float v = alpha * acc[i][j][r] + bv;
            if (relu) v = fmaxf(v, 0.0f);
            if (Cf) Cf[(size_t)row * N + col] = v;
            if (Ch) Ch[(size_t)row * ldch + col] = (_Float16)v;
          } else if (Ch && col < ldch) {
            Ch[(size_t)row * ldch + col] = (_Float16)0.0f;  // keep pad zeroed
          }
        }
      }
    }
  }
}

// ---------------------------------------------------------------------------
// Gate: p = l2norm(softmax(leaky_relu(concat(parts) @ W + b)))  (ncols <= 5)
// One block per row, 256-thread K reduction.
// ---------------------------------------------------------------------------
__global__ __launch_bounds__(256) void gate_kernel(
    const float* q0, int k0, const float* q1, int k1, const float* q2, int k2,
    const float* q3, int k3, const float* q4, int k4, int nparts,
    const float* __restrict__ W, const float* __restrict__ b,
    float* __restrict__ out, int ncols, int rows)
{
  const int r = blockIdx.x;
  const int tid = threadIdx.x;
  const float* qs[5] = {q0, q1, q2, q3, q4};
  const int    ks[5] = {k0, k1, k2, k3, k4};
  float acc[5] = {0.f, 0.f, 0.f, 0.f, 0.f};
  int koff = 0;
  for (int p = 0; p < nparts; ++p) {
    const float* src = qs[p] + (size_t)r * ks[p];
    for (int k = tid; k < ks[p]; k += 256) {
      const float v = src[k];
      const float* wr = W + (size_t)(koff + k) * ncols;
      for (int c = 0; c < ncols; ++c) acc[c] += v * wr[c];
    }
    koff += ks[p];
  }
  __shared__ float red[256];
  __shared__ float colv[5];
  for (int c = 0; c < ncols; ++c) {
    red[tid] = acc[c];
    __syncthreads();
    for (int s = 128; s > 0; s >>= 1) {
      if (tid < s) red[tid] += red[tid + s];
      __syncthreads();
    }
    if (tid == 0) colv[c] = red[0];
    __syncthreads();
  }
  if (tid == 0 && r < rows) {
    float y[5]; float mx = -1e30f;
    for (int c = 0; c < ncols; ++c) {
      float v = colv[c] + b[c];
      v = (v > 0.0f) ? v : 0.01f * v;           // leaky_relu(0.01)
      y[c] = v; mx = fmaxf(mx, v);
    }
    float s = 0.0f;
    for (int c = 0; c < ncols; ++c) { y[c] = __expf(y[c] - mx); s += y[c]; }
    float n2 = 0.0f;
    for (int c = 0; c < ncols; ++c) { y[c] /= s; n2 += y[c] * y[c]; }
    const float nn = fmaxf(sqrtf(n2), 1e-12f);
    for (int c = 0; c < ncols; ++c) out[(size_t)r * ncols + c] = y[c] / nn;
  }
}

// m = p[:,0]*za + p[:,1]*hb  -> f16, padded ld (pad zero-filled)
__global__ void mix_kernel(const float* __restrict__ p,
                           const float* __restrict__ za,
                           const float* __restrict__ hb,
                           _Float16* __restrict__ out, int rows, int cols,
                           int ldo)
{
  const int i = blockIdx.x * blockDim.x + threadIdx.x;
  if (i >= rows * ldo) return;
  const int r = i / ldo;
  const int c = i - r * ldo;
  float v = 0.0f;
  if (c < cols) {
    const size_t s = (size_t)r * cols + c;
    v = p[(size_t)r * 2 + 0] * za[s] + p[(size_t)r * 2 + 1] * hb[s];
  }
  out[i] = (_Float16)v;
}

// out[r, :] = concat_i( w[r,i] * parts[i][r, :] ) -> f16, padded ld
__global__ void scale_concat_kernel(
    const float* __restrict__ w, int nw,
    const float* q0, int k0, const float* q1, int k1, const float* q2, int k2,
    const float* q3, int k3, const float* q4, int k4, int nparts,
    _Float16* __restrict__ out, int rows, int ktot, int ldo)
{
  const int i = blockIdx.x * blockDim.x + threadIdx.x;
  if (i >= rows * ldo) return;
  const int r = i / ldo;
  int cc = i - r * ldo;
  if (cc >= ktot) { out[i] = (_Float16)0.0f; return; }
  const float* qs[5] = {q0, q1, q2, q3, q4};
  const int    ks[5] = {k0, k1, k2, k3, k4};
  int p = 0;
  while (p < nparts - 1 && cc >= ks[p]) { cc -= ks[p]; ++p; }
  const float v = w[(size_t)r * nw + p] * qs[p][(size_t)r * ks[p] + cc];
  out[i] = (_Float16)v;
}

// f32[K,N] -> f16[K,ldo] with zero-filled pad cols
__global__ void cvt_pad_kernel(const float* __restrict__ src,
                               _Float16* __restrict__ dst, int K, int N,
                               int ldo)
{
  const int i = blockIdx.x * blockDim.x + threadIdx.x;
  if (i >= K * ldo) return;
  const int k = i / ldo;
  const int n = i - k * ldo;
  dst[i] = (n < N) ? (_Float16)src[(size_t)k * N + n] : (_Float16)0.0f;
}

__global__ void softmax10_kernel(const float* __restrict__ in,
                                 float* __restrict__ out, int rows)
{
  const int r = blockIdx.x * blockDim.x + threadIdx.x;
  if (r >= rows) return;
  float mx = -1e30f;
  for (int c = 0; c < 10; ++c) mx = fmaxf(mx, in[(size_t)r * 10 + c]);
  float e[10], s = 0.0f;
  for (int c = 0; c < 10; ++c) { e[c] = __expf(in[(size_t)r * 10 + c] - mx); s += e[c]; }
  for (int c = 0; c < 10; ++c) out[(size_t)r * 10 + c] = e[c] / s;
}

// Student-t assignment, V=1:  q = (1+||z-mu||^2)^-1, row-normalized.
__global__ void q_kernel(const float* __restrict__ z,
                         const float* __restrict__ mu,
                         float* __restrict__ q, int rows)
{
  const int r = blockIdx.x * blockDim.x + threadIdx.x;
  if (r >= rows) return;
  float zz[10];
  for (int k = 0; k < 10; ++k) zz[k] = z[(size_t)r * 10 + k];
  float qq[10], s = 0.0f;
  for (int c = 0; c < 10; ++c) {
    float d2 = 0.0f;
    for (int k = 0; k < 10; ++k) {
      const float d = zz[k] - mu[c * 10 + k];
      d2 += d * d;
    }
    qq[c] = 1.0f / (1.0f + d2);
    s += qq[c];
  }
  for (int c = 0; c < 10; ++c) q[(size_t)r * 10 + c] = qq[c] / s;
}

// ---------------------------------------------------------------------------
// Host orchestration
// ---------------------------------------------------------------------------
static void launch_gemm(hipStream_t s, bool a16, bool relu, const void* A,
                        int lda, const _Float16* B, int ldb, const float* bias,
                        float* Cf, _Float16* Ch, int ldch, int M, int N, int K,
                        float alpha, float ascale)
{
  dim3 grid((M + BM - 1) / BM, (N + BN - 1) / BN);
  dim3 blk(256);
  if (a16)
    gemm_wmma<true><<<grid, blk, 0, s>>>(A, lda, B, ldb, bias, Cf, Ch, ldch,
                                         M, N, K, alpha, ascale, relu ? 1 : 0);
  else
    gemm_wmma<false><<<grid, blk, 0, s>>>(A, lda, B, ldb, bias, Cf, Ch, ldch,
                                          M, N, K, alpha, ascale, relu ? 1 : 0);
}

extern "C" void kernel_launch(void* const* d_in, const int* in_sizes, int n_in,
                              void* d_out, int out_size, void* d_ws, size_t ws_size,
                              hipStream_t stream)
{
  (void)in_sizes; (void)n_in; (void)out_size; (void)ws_size;
  const int Nn = 6000;
  const float ADJ_S = 8192.0f;            // keep f16 adj entries normal
  const float INV_S = 1.0f / 8192.0f;

  const float* x = (const float*)d_in[0];
  const float* adjs[4] = {(const float*)d_in[1], (const float*)d_in[2],
                          (const float*)d_in[3], (const float*)d_in[4]};
  const float* be1 = (const float*)d_in[6];
  const float* be2 = (const float*)d_in[8];
  const float* be3 = (const float*)d_in[10];
  const float* bz  = (const float*)d_in[12];
  const float* bd1 = (const float*)d_in[14];
  const float* bd2 = (const float*)d_in[16];
  const float* bd3 = (const float*)d_in[18];
  const float* bxb = (const float*)d_in[20];
  const float* W1 = (const float*)d_in[27]; const float* b1 = (const float*)d_in[28];
  const float* W2 = (const float*)d_in[29]; const float* b2 = (const float*)d_in[30];
  const float* W3 = (const float*)d_in[31]; const float* b3 = (const float*)d_in[32];
  const float* Wl = (const float*)d_in[33]; const float* bl = (const float*)d_in[34];
  const float* Wn = (const float*)d_in[35]; const float* bn = (const float*)d_in[36];
  const float* mu = (const float*)d_in[37];

  // ---- workspace arena ----
  char* base = (char*)d_ws;
  size_t off = 0;
  auto alloc  = [&](size_t bytes) -> void* {
    void* p = base + off;
    off = (off + bytes + 255) & ~(size_t)255;
    return p;
  };
  auto allocH = [&](size_t n) { return (_Float16*)alloc(n * sizeof(_Float16)); };
  auto allocF = [&](size_t n) { return (float*)alloc(n * sizeof(float)); };

  // f16 weight copies (padded leading dims, pad zero-filled)
  _Float16* We1h = allocH((size_t)1000 * 504);
  _Float16* We2h = allocH((size_t)500 * 504);
  _Float16* We3h = allocH((size_t)500 * 2000);
  _Float16* Wzh  = allocH((size_t)2000 * 16);
  _Float16* Wd1h = allocH((size_t)10 * 2000);
  _Float16* Wd2h = allocH((size_t)2000 * 504);
  _Float16* Wd3h = allocH((size_t)500 * 504);
  _Float16* Wxbh = allocH((size_t)500 * 1000);
  _Float16* Wg0h = allocH((size_t)1000 * 504);
  _Float16* Wg1h = allocH((size_t)500 * 504);
  _Float16* Wg2h = allocH((size_t)500 * 2000);
  _Float16* Wg3h = allocH((size_t)2000 * 16);
  _Float16* Wgzh = allocH((size_t)3020 * 16);
  _Float16* Wgnh = allocH((size_t)40 * 16);

  // f16 activation buffers
  _Float16* pp0    = allocH((size_t)Nn * 2000);   // AE ping
  _Float16* pp1    = allocH((size_t)Nn * 2000);   // AE pong
  _Float16* t0_h   = allocH((size_t)Nn * 504);    // x @ Wg0 (branch invariant)
  _Float16* tbuf_h = allocH((size_t)Nn * 2000);   // feat @ W, consumed by prop
  _Float16* mix_h  = allocH((size_t)Nn * 2000);
  _Float16* net_h  = allocH((size_t)Nn * 3024);
  _Float16* nfin_h = allocH((size_t)Nn * 40);

  // f32 buffers
  float* h1  = allocF((size_t)Nn * 500);
  float* h2  = allocF((size_t)Nn * 500);
  float* h3  = allocF((size_t)Nn * 2000);
  float* zf  = allocF((size_t)Nn * 10);
  float* z1  = allocF((size_t)Nn * 500);
  float* z2  = allocF((size_t)Nn * 500);
  float* z3  = allocF((size_t)Nn * 2000);
  float* z4  = allocF((size_t)Nn * 10);
  float* nets = allocF((size_t)4 * Nn * 10);      // nets[i] = nets + i*60000
  float* pg  = allocF((size_t)Nn * 5);            // gate scratch
  float* wnf = allocF((size_t)Nn * 4);

  float* out       = (float*)d_out;
  float* out_xbar  = out;                 // 6000 x 1000
  float* out_q     = out + 6000000;       // 6000 x 10
  float* out_pred  = out + 6060000;       // 6000 x 10
  float* out_netf  = out + 6120000;       // 6000 x 10

  auto cvt = [&](int idx, _Float16* dst, int K, int N, int ld) {
    const int cnt = K * ld;
    cvt_pad_kernel<<<dim3((cnt + 255) / 256), dim3(256), 0, stream>>>(
        (const float*)d_in[idx], dst, K, N, ld);
  };
  cvt(5,  We1h, 1000, 500, 504);   cvt(7,  We2h, 500, 500, 504);
  cvt(9,  We3h, 500, 2000, 2000);  cvt(11, Wzh,  2000, 10, 16);
  cvt(13, Wd1h, 10, 2000, 2000);   cvt(15, Wd2h, 2000, 500, 504);
  cvt(17, Wd3h, 500, 500, 504);    cvt(19, Wxbh, 500, 1000, 1000);
  cvt(21, Wg0h, 1000, 500, 504);   cvt(22, Wg1h, 500, 500, 504);
  cvt(23, Wg2h, 500, 2000, 2000);  cvt(24, Wg3h, 2000, 10, 16);
  cvt(25, Wgzh, 3020, 10, 16);     cvt(26, Wgnh, 40, 10, 16);

  auto gate = [&](const float* q0, int k0, const float* q1, int k1,
                  const float* q2, int k2, const float* q3, int k3,
                  const float* q4, int k4, int np, const float* W,
                  const float* b, float* o, int nc) {
    gate_kernel<<<dim3(Nn), dim3(256), 0, stream>>>(
        q0, k0, q1, k1, q2, k2, q3, k3, q4, k4, np, W, b, o, nc, Nn);
  };
  auto mix = [&](const float* p, const float* za, const float* hb,
                 _Float16* o, int cols, int ldo) {
    const int cnt = Nn * ldo;
    mix_kernel<<<dim3((cnt + 255) / 256), dim3(256), 0, stream>>>(
        p, za, hb, o, Nn, cols, ldo);
  };

  // ---------------- autoencoder ----------------
  launch_gemm(stream, false, true,  x,   1000, We1h, 504,  be1, h1, pp0, 504,  Nn, 500,  1000, 1.f, 1.f);
  launch_gemm(stream, true,  true,  pp0, 504,  We2h, 504,  be2, h2, pp1, 504,  Nn, 500,  500,  1.f, 1.f);
  launch_gemm(stream, true,  true,  pp1, 504,  We3h, 2000, be3, h3, pp0, 2000, Nn, 2000, 500,  1.f, 1.f);
  launch_gemm(stream, true,  false, pp0, 2000, Wzh,  16,   bz,  zf, pp1, 16,   Nn, 10,   2000, 1.f, 1.f);
  launch_gemm(stream, true,  true,  pp1, 16,   Wd1h, 2000, bd1, nullptr, pp0, 2000, Nn, 2000, 10,   1.f, 1.f);
  launch_gemm(stream, true,  true,  pp0, 2000, Wd2h, 504,  bd2, nullptr, pp1, 504,  Nn, 500,  2000, 1.f, 1.f);
  launch_gemm(stream, true,  true,  pp1, 504,  Wd3h, 504,  bd3, nullptr, pp0, 504,  Nn, 500,  500,  1.f, 1.f);
  launch_gemm(stream, true,  false, pp0, 504,  Wxbh, 1000, bxb, out_xbar, nullptr, 0, Nn, 1000, 500, 1.f, 1.f);

  // t0 = x @ Wg0 (shared by all branches)
  launch_gemm(stream, false, false, x, 1000, Wg0h, 504, nullptr, nullptr, t0_h, 504, Nn, 500, 1000, 1.f, 1.f);

  // ---------------- 4 GNN branches ----------------
  for (int bi = 0; bi < 4; ++bi) {
    const float* adj = adjs[bi];
    float* net_out = nets + (size_t)bi * Nn * 10;

    // z1 = relu(adj @ t0)
    launch_gemm(stream, false, true, adj, Nn, t0_h, 504, nullptr, z1, nullptr, 0, Nn, 500, Nn, INV_S, ADJ_S);
    gate(h1, 500, z1, 500, nullptr, 0, nullptr, 0, nullptr, 0, 2, W1, b1, pg, 2);
    mix(pg, z1, h1, mix_h, 500, 504);
    launch_gemm(stream, true, false, mix_h, 504, Wg1h, 504, nullptr, nullptr, tbuf_h, 504, Nn, 500, 500, 1.f, 1.f);
    // z2 = relu(adj @ (m1 @ Wg1))
    launch_gemm(stream, false, true, adj, Nn, tbuf_h, 504, nullptr, z2, nullptr, 0, Nn, 500, Nn, INV_S, ADJ_S);
    gate(h2, 500, z2, 500, nullptr, 0, nullptr, 0, nullptr, 0, 2, W2, b2, pg, 2);
    mix(pg, z2, h2, mix_h, 500, 504);
    launch_gemm(stream, true, false, mix_h, 504, Wg2h, 2000, nullptr, nullptr, tbuf_h, 2000, Nn, 2000, 500, 1.f, 1.f);
    // z3 = relu(adj @ (m2 @ Wg2))
    launch_gemm(stream, false, true, adj, Nn, tbuf_h, 2000, nullptr, z3, nullptr, 0, Nn, 2000, Nn, INV_S, ADJ_S);
    gate(h3, 2000, z3, 2000, nullptr, 0, nullptr, 0, nullptr, 0, 2, W3, b3, pg, 2);
    mix(pg, z3, h3, mix_h, 2000, 2000);
    launch_gemm(stream, true, false, mix_h, 2000, Wg3h, 16, nullptr, nullptr, tbuf_h, 16, Nn, 10, 2000, 1.f, 1.f);
    // z4 = relu(adj @ (m3 @ Wg3))
    launch_gemm(stream, false, true, adj, Nn, tbuf_h, 16, nullptr, z4, nullptr, 0, Nn, 10, Nn, INV_S, ADJ_S);
    // w = gate([z1,z2,z3,z4,z])
    gate(z1, 500, z2, 500, z3, 2000, z4, 10, zf, 10, 5, Wl, bl, pg, 5);
    {
      const int cnt = Nn * 3024;
      scale_concat_kernel<<<dim3((cnt + 255) / 256), dim3(256), 0, stream>>>(
          pg, 5, z1, 500, z2, 500, z3, 2000, z4, 10, zf, 10, 5, net_h, Nn, 3020, 3024);
    }
    launch_gemm(stream, true, false, net_h, 3024, Wgzh, 16, nullptr, nullptr, tbuf_h, 16, Nn, 10, 3020, 1.f, 1.f);
    // net = adj @ (net_cat @ Wgz)    (no activation)
    launch_gemm(stream, false, false, adj, Nn, tbuf_h, 16, nullptr, net_out, nullptr, 0, Nn, 10, Nn, INV_S, ADJ_S);
  }

  // ---------------- fusion head ----------------
  const float* n0p = nets;
  const float* n1p = nets + (size_t)Nn * 10;
  const float* n2p = nets + (size_t)2 * Nn * 10;
  const float* n3p = nets + (size_t)3 * Nn * 10;
  gate(n0p, 10, n1p, 10, n2p, 10, n3p, 10, nullptr, 0, 4, Wn, bn, wnf, 4);
  {
    const int cnt = Nn * 40;
    scale_concat_kernel<<<dim3((cnt + 255) / 256), dim3(256), 0, stream>>>(
        wnf, 4, n0p, 10, n1p, 10, n2p, 10, n3p, 10, nullptr, 0, 4, nfin_h, Nn, 40, 40);
  }
  launch_gemm(stream, true, false, nfin_h, 40, Wgnh, 16, nullptr, nullptr, tbuf_h, 16, Nn, 10, 40, 1.f, 1.f);
  // netf = adj1 @ (netf_in @ Wgn)
  launch_gemm(stream, false, false, adjs[0], Nn, tbuf_h, 16, nullptr, out_netf, nullptr, 0, Nn, 10, Nn, INV_S, ADJ_S);

  softmax10_kernel<<<dim3((Nn + 255) / 256), dim3(256), 0, stream>>>(out_netf, out_pred, Nn);
  q_kernel<<<dim3((Nn + 255) / 256), dim3(256), 0, stream>>>(zf, mu, out_q, Nn);
}
